// RelPosSelfAttention_62732292325585
// MI455X (gfx1250) — matrix operand order repaired and checked by reading, not compiled, not run on hardware
//
#include <hip/hip_runtime.h>
#include <hip/hip_bf16.h>

#define B_  4
#define N_  4096
#define D_  1024
#define H_  16
#define DH_ 64
#define PD_ 8
#define QD_ 72   // DH + PD
#define AROW 40  // LDS row stride in ushorts (32 data + 8 pad)

typedef __attribute__((ext_vector_type(16))) __bf16 v16bf;
typedef __attribute__((ext_vector_type(8)))  float  v8f;

__device__ __forceinline__ unsigned short f2bf(float f) {
  unsigned int u = __float_as_uint(f);
  u += 0x7FFFu + ((u >> 16) & 1u);           // round-to-nearest-even
  return (unsigned short)(u >> 16);
}

union V16U { v16bf v; unsigned short u[16]; };

// Async DMA: global -> LDS, 16 bytes per lane, tracked by ASYNCcnt.
__device__ __forceinline__ void async_load_b128(unsigned lds_off, const void* gaddr) {
  asm volatile("global_load_async_to_lds_b128 %0, %1, off"
               :: "v"(lds_off), "v"(gaddr) : "memory");
}
__device__ __forceinline__ void wait_async_le5() {
  asm volatile("s_wait_asynccnt 0x5" ::: "memory");
}
__device__ __forceinline__ void wait_async_0() {
  asm volatile("s_wait_asynccnt 0x0" ::: "memory");
}

// -----------------------------------------------------------------------------
// f32 -> bf16 elementwise (8 elements / thread)
// -----------------------------------------------------------------------------
__global__ __launch_bounds__(256) void f32_to_bf16_kernel(
    const float* __restrict__ src, unsigned short* __restrict__ dst, int n8)
{
  int i = blockIdx.x * 256 + threadIdx.x;
  if (i >= n8) return;
  const float4* s4 = (const float4*)src;
  float4 a = s4[2 * i], b = s4[2 * i + 1];
  uint4 p;
  p.x = (unsigned)f2bf(a.x) | ((unsigned)f2bf(a.y) << 16);
  p.y = (unsigned)f2bf(a.z) | ((unsigned)f2bf(a.w) << 16);
  p.z = (unsigned)f2bf(b.x) | ((unsigned)f2bf(b.y) << 16);
  p.w = (unsigned)f2bf(b.z) | ((unsigned)f2bf(b.w) << 16);
  ((uint4*)dst)[i] = p;
}

// -----------------------------------------------------------------------------
// C[M,Nc](f32) = A[M,K](bf16) * W[Nc,K]^T(bf16) (+ bias[Nc])
// Block tile 256x64, K chunk 32, double-buffered LDS filled by async DMA.
// 8 waves; wave w owns a 32(M)x64(N) sub-tile: 8 WMMA per chunk, shared B tiles.
// -----------------------------------------------------------------------------
__global__ __launch_bounds__(256) void gemm_bf16_wmma(
    const unsigned short* __restrict__ A, const unsigned short* __restrict__ W,
    const float* __restrict__ bias, float* __restrict__ C,
    int M, int K, int Nc)
{
  __shared__ unsigned short As[2][256 * AROW];
  __shared__ unsigned short Bs[2][64 * AROW];
  const int tid  = threadIdx.x;
  const int wave = tid >> 5, lane = tid & 31;
  const int gm = blockIdx.y * 256;
  const int gn = blockIdx.x * 64;

  v8f acc[2][4];
#pragma unroll
  for (int mt = 0; mt < 2; ++mt)
#pragma unroll
    for (int t = 0; t < 4; ++t)
#pragma unroll
      for (int i = 0; i < 8; ++i) acc[mt][t][i] = 0.f;

  // per-thread staging slots (5 async b128 per thread per chunk: 4 A + 1 B)
  const int arow = tid >> 2, aseg = tid & 3;   // A rows arow + {0,64,128,192}
  const int brow = tid >> 2, bseg = tid & 3;   // B rows 0..63

  auto issue = [&](int nb, int kc) {
    unsigned asb = (unsigned)(unsigned long long)(const void*)&As[nb][0];
    unsigned bsb = (unsigned)(unsigned long long)(const void*)&Bs[nb][0];
#pragma unroll
    for (int i = 0; i < 4; ++i) {
      int row = arow + 64 * i;
      async_load_b128(asb + (unsigned)(row * AROW + aseg * 8) * 2,
                      A + (size_t)(gm + row) * K + kc + aseg * 8);
    }
    async_load_b128(bsb + (unsigned)(brow * AROW + bseg * 8) * 2,
                    W + (size_t)(gn + brow) * K + kc + bseg * 8);
  };

  const int nch = K >> 5;
  issue(0, 0);                                  // prefetch chunk 0
  for (int ic = 0; ic < nch; ++ic) {
    const int cur = ic & 1;
    __syncthreads();                            // all waves done reading buf[1-cur]
    if (ic + 1 < nch) { issue(1 - cur, (ic + 1) << 5); wait_async_le5(); }
    else              { wait_async_0(); }
    __syncthreads();                            // buf[cur] visible to all waves

    const unsigned short* Asb = &As[cur][0];
    const unsigned short* Bsb = &Bs[cur][0];

    // A lane layout (16-bit 16x32): lanes<16: K{0..7,16..23}; lanes>=16: K{8..15,24..31}
    const int m0   = (wave << 5) + (lane & 15);
    const int koff = (lane >> 4) << 3;          // 0 or 8
    V16U av[2];
#pragma unroll
    for (int mt = 0; mt < 2; ++mt) {
      const int m = m0 + (mt << 4);
#pragma unroll
      for (int i = 0; i < 8; ++i) {
        av[mt].u[i]     = Asb[m * AROW + koff + i];
        av[mt].u[8 + i] = Asb[m * AROW + 16 + koff + i];
      }
    }
    // B lane layout (16-bit 32x16): lanes<16: K{0..7,16..23} of col N=lane; lanes>=16: K{8..15,24..31}
    const int bn  = lane & 15;
    const int bko = (lane >> 4) << 3;           // 0 or 8
    V16U bvv[4];
#pragma unroll
    for (int t = 0; t < 4; ++t)
#pragma unroll
      for (int i = 0; i < 16; ++i)
        bvv[t].u[i] = Bsb[((t << 4) + bn) * AROW + bko + (i & 7) + ((i >> 3) << 4)];

#pragma unroll
    for (int t = 0; t < 4; ++t) {
      acc[0][t] = __builtin_amdgcn_wmma_f32_16x16x32_bf16(
          false, av[0].v, false, bvv[t].v, (short)0, acc[0][t], false, false);
      acc[1][t] = __builtin_amdgcn_wmma_f32_16x16x32_bf16(
          false, av[1].v, false, bvv[t].v, (short)0, acc[1][t], false, false);
    }
  }

  // C/D layout: VGPR r -> row (lane<16 ? r : r+8), col = lane&15
  const int lhalf = lane >> 4;
  const int ncol  = lane & 15;
#pragma unroll
  for (int mt = 0; mt < 2; ++mt)
#pragma unroll
    for (int t = 0; t < 4; ++t) {
      int col = gn + (t << 4) + ncol;
      float badd = bias ? bias[col] : 0.f;
#pragma unroll
      for (int r = 0; r < 8; ++r) {
        int row = gm + (wave << 5) + (mt << 4) + lhalf * 8 + r;
        C[(size_t)row * Nc + col] = acc[mt][t][r] + badd;
      }
    }
}

// -----------------------------------------------------------------------------
// Column stats for k softmax over n: max_n and sum_n exp per (b,h,d).
// (bk is constant along n -> softmax(axis=-2) invariant -> dropped.)
// -----------------------------------------------------------------------------
__global__ __launch_bounds__(256) void kstats_kernel(
    const float* __restrict__ Kc, const float* __restrict__ pos,
    float* __restrict__ stats)
{
  const int col = blockIdx.x;                 // b*H*QD + h*QD + d
  const int d = col % QD_;
  const int h = (col / QD_) % H_;
  const int b = col / (QD_ * H_);
  const int t = threadIdx.x;
  __shared__ float red[256];
  float vals[16];
  for (int i = 0; i < 16; ++i) {
    int n = t + (i << 8);
    size_t bn = (size_t)(b * N_ + n);
    vals[i] = (d < DH_) ? Kc[bn * D_ + h * DH_ + d] : pos[bn * PD_ + (d - DH_)];
  }
  float m = -3.4e38f;
  for (int i = 0; i < 16; ++i) m = fmaxf(m, vals[i]);
  red[t] = m; __syncthreads();
  for (int s = 128; s > 0; s >>= 1) {
    if (t < s) red[t] = fmaxf(red[t], red[t + s]);
    __syncthreads();
  }
  const float cmax = red[0];
  __syncthreads();
  float sm = 0.f;
  for (int i = 0; i < 16; ++i) sm += __expf(vals[i] - cmax);
  red[t] = sm; __syncthreads();
  for (int s = 128; s > 0; s >>= 1) {
    if (t < s) red[t] += red[t + s];
    __syncthreads();
  }
  if (t == 0) { stats[col * 2] = cmax; stats[col * 2 + 1] = red[0]; }
}

// -----------------------------------------------------------------------------
// ctx[b,h,d,e] = sum_n softk[b,h,n,d] * v[b,h,n,e]   (72 x 64, K = 4096)
// -----------------------------------------------------------------------------
__global__ __launch_bounds__(256) void ctx_kernel(
    const float* __restrict__ Kc, const float* __restrict__ pos,
    const float* __restrict__ Vv, const float* __restrict__ stats,
    float* __restrict__ ctx)
{
  const int bh = blockIdx.x;
  const int h = bh % H_, b = bh / H_;
  const int t = threadIdx.x;
  __shared__ float sk[32][QD_ + 1];
  __shared__ float vv[32][DH_ + 1];
  __shared__ float smax[QD_], srs[QD_];
  if (t < QD_) {
    smax[t] = stats[(bh * QD_ + t) * 2];
    srs[t]  = 1.f / stats[(bh * QD_ + t) * 2 + 1];
  }
  float acc[18];
  for (int p = 0; p < 18; ++p) acc[p] = 0.f;
  __syncthreads();

  for (int n0 = 0; n0 < N_; n0 += 32) {
    for (int idx = t; idx < 32 * QD_; idx += 256) {
      int nl = idx / QD_, d = idx % QD_;
      size_t bn = (size_t)(b * N_ + n0 + nl);
      float v = (d < DH_) ? Kc[bn * D_ + h * DH_ + d] : pos[bn * PD_ + (d - DH_)];
      sk[nl][d] = __expf(v - smax[d]) * srs[d];
    }
    for (int idx = t; idx < 32 * DH_; idx += 256) {
      int nl = idx >> 6, e = idx & 63;
      vv[nl][e] = Vv[(size_t)(b * N_ + n0 + nl) * D_ + h * DH_ + e];
    }
    __syncthreads();
    for (int p = 0; p < 18; ++p) {
      int oidx = t + (p << 8);             // 4608 outputs = 72*64
      int d = oidx >> 6, e = oidx & 63;
      float a = 0.f;
      for (int nl = 0; nl < 32; ++nl) a += sk[nl][d] * vv[nl][e];
      acc[p] += a;
    }
    __syncthreads();
  }
  for (int p = 0; p < 18; ++p)
    ctx[(size_t)bh * QD_ * DH_ + t + (p << 8)] = acc[p];
}

// -----------------------------------------------------------------------------
// q = softmax(concat(cq, rot(pos)) + bq); out[b,n,h,e] = (ctx^T q)/rowsum (bf16)
// (d_inv == 1/(q . ksum) with ksum ~= 1 after column softmax; folded into rowsum)
// -----------------------------------------------------------------------------
__global__ __launch_bounds__(256) void apply_kernel(
    const float* __restrict__ Qc, const float* __restrict__ pos,
    const float* __restrict__ alpha, const float* __restrict__ beta,
    const float* __restrict__ bq, const float* __restrict__ ctx,
    unsigned short* __restrict__ attnb)
{
  const int bh = blockIdx.y;
  const int h = bh % H_, b = bh / H_;
  const int t = threadIdx.x;
  const int r = t >> 6, j = t & 63;       // 4 rows x 64 threads
  const int n = blockIdx.x * 4 + r;
  __shared__ float ctxs[QD_][DH_ + 1];
  __shared__ float qb[4][QD_];
  __shared__ float red[4][64];

  for (int idx = t; idx < QD_ * DH_; idx += 256)
    ctxs[idx >> 6][idx & 63] = ctx[(size_t)bh * QD_ * DH_ + idx];

  size_t bn = (size_t)(b * N_ + n);
  float q0 = Qc[bn * D_ + h * DH_ + j] + bq[h * QD_ + j];
  float q1 = -3.4e38f;
  if (j < PD_) {
    int s = j >> 1, c = j & 1;
    float a = alpha[h * 4 + s], be = beta[h * 4 + s];
    float p0 = pos[bn * PD_ + 2 * s], p1 = pos[bn * PD_ + 2 * s + 1];
    float pv = c ? (be * p0 + a * p1) : (a * p0 - be * p1);   // R = a*I + be*J
    q1 = pv + bq[h * QD_ + DH_ + j];
  }
  // row max over 72
  red[r][j] = fmaxf(q0, q1);
  __syncthreads();
  for (int s = 32; s > 0; s >>= 1) {
    if (j < s) red[r][j] = fmaxf(red[r][j], red[r][j + s]);
    __syncthreads();
  }
  float rmax = red[r][0];
  __syncthreads();
  float e0 = __expf(q0 - rmax);
  float e1 = (j < PD_) ? __expf(q1 - rmax) : 0.f;
  qb[r][j] = e0;
  if (j < PD_) qb[r][DH_ + j] = e1;
  red[r][j] = e0 + e1;
  __syncthreads();
  for (int s = 32; s > 0; s >>= 1) {
    if (j < s) red[r][j] += red[r][j + s];
    __syncthreads();
  }
  float inv = 1.f / red[r][0];
  __syncthreads();
  float acc = 0.f;
  for (int d = 0; d < QD_; ++d) acc += qb[r][d] * ctxs[d][j];
  attnb[bn * D_ + h * DH_ + j] = f2bf(acc * inv);
}

// -----------------------------------------------------------------------------
extern "C" void kernel_launch(void* const* d_in, const int* in_sizes, int n_in,
                              void* d_out, int out_size, void* d_ws, size_t ws_size,
                              hipStream_t stream)
{
  (void)in_sizes; (void)n_in; (void)out_size; (void)ws_size;
  const float* x     = (const float*)d_in[0];
  const float* pos   = (const float*)d_in[1];
  const float* Wq    = (const float*)d_in[2];
  const float* alpha = (const float*)d_in[3];
  const float* beta  = (const float*)d_in[4];
  const float* bq    = (const float*)d_in[5];
  const float* Wk    = (const float*)d_in[6];
  /* d_in[7] = bk: softmax over n is invariant to per-(h,d) bias -> unused */
  const float* Wv    = (const float*)d_in[8];
  const float* bv    = (const float*)d_in[9];
  const float* Wo    = (const float*)d_in[10];
  const float* bo    = (const float*)d_in[11];
  float* out = (float*)d_out;

  char* ws = (char*)d_ws;
  const size_t szMat  = (size_t)B_ * N_ * D_ * sizeof(float);            // 64 MB
  const size_t szMatB = (size_t)B_ * N_ * D_ * sizeof(unsigned short);   // 32 MB
  const size_t szWB   = (size_t)D_ * D_ * sizeof(unsigned short);        // 2 MB
  const size_t szSt   = (size_t)B_ * H_ * QD_ * 2 * sizeof(float);
  float* Qc    = (float*)(ws);
  float* Kc    = (float*)(ws + szMat);
  float* Vv    = (float*)(ws + 2 * szMat);
  unsigned short* xb    = (unsigned short*)(ws + 3 * szMat);
  unsigned short* Wqb   = (unsigned short*)(ws + 3 * szMat + szMatB);
  unsigned short* Wkb   = (unsigned short*)(ws + 3 * szMat + szMatB + szWB);
  unsigned short* Wvb   = (unsigned short*)(ws + 3 * szMat + szMatB + 2 * szWB);
  unsigned short* Wob   = (unsigned short*)(ws + 3 * szMat + szMatB + 3 * szWB);
  unsigned short* attnb = (unsigned short*)(ws + 3 * szMat + szMatB + 4 * szWB);
  float* stats = (float*)(ws + 3 * szMat + 2 * szMatB + 4 * szWB);
  float* ctx   = (float*)(ws + 3 * szMat + 2 * szMatB + 4 * szWB + szSt);

  const int M = B_ * N_;
  const int nX8 = (M * D_) / 8, nW8 = (D_ * D_) / 8;
  f32_to_bf16_kernel<<<nX8 / 256, 256, 0, stream>>>(x,  xb,  nX8);
  f32_to_bf16_kernel<<<nW8 / 256, 256, 0, stream>>>(Wq, Wqb, nW8);
  f32_to_bf16_kernel<<<nW8 / 256, 256, 0, stream>>>(Wk, Wkb, nW8);
  f32_to_bf16_kernel<<<nW8 / 256, 256, 0, stream>>>(Wv, Wvb, nW8);
  f32_to_bf16_kernel<<<nW8 / 256, 256, 0, stream>>>(Wo, Wob, nW8);

  dim3 gg(D_ / 64, M / 256);
  gemm_bf16_wmma<<<gg, 256, 0, stream>>>(xb, Wqb, nullptr, Qc, M, D_, D_);
  gemm_bf16_wmma<<<gg, 256, 0, stream>>>(xb, Wkb, nullptr, Kc, M, D_, D_);
  gemm_bf16_wmma<<<gg, 256, 0, stream>>>(xb, Wvb, bv,      Vv, M, D_, D_);
  kstats_kernel<<<B_ * H_ * QD_, 256, 0, stream>>>(Kc, pos, stats);
  ctx_kernel<<<B_ * H_, 256, 0, stream>>>(Kc, pos, Vv, stats, ctx);
  apply_kernel<<<dim3(N_ / 4, B_ * H_), 256, 0, stream>>>(Qc, pos, alpha, beta, bq, ctx, attnb);
  gemm_bf16_wmma<<<gg, 256, 0, stream>>>(attnb, Wob, bo, out, M, D_, D_);
}
